// Conv2dfft_10445360463935
// MI455X (gfx1250) — matrix operand order repaired and checked by reading, not compiled, not run on hardware
//
#include <hip/hip_runtime.h>

typedef __attribute__((ext_vector_type(16))) __bf16       v16bf;
typedef __attribute__((ext_vector_type(8)))  float        v8f;
typedef __attribute__((ext_vector_type(4)))  unsigned int u32x4;

#define N_IMG 32
#define C_IN  128
#define F_OUT 128
#define H_IMG 32
#define W_IMG 32
#define K_TOT (C_IN * 9)              // 1152
#define CP    136                     // padded channel pitch (elems): 68 dwords == 4 mod 64 -> bank-tiled
#define SLAB_ELEMS (3 * 34 * CP)      // 13872 bf16 (~27.1 KB)
#define XB_ELEMS (N_IMG * C_IN * H_IMG * W_IMG)  // 4194304
#define WB_ELEMS (F_OUT * K_TOT)                 // 147456

__device__ __forceinline__ unsigned short f32_to_bf16_rne(float f) {
  unsigned int u = __float_as_uint(f);
  u += 0x7FFFu + ((u >> 16) & 1u);
  return (unsigned short)(u >> 16);
}

// Prep: x (NCHW fp32) -> xt (NHWC bf16); w (F,C,3,3 fp32) -> wb[f][k'] bf16
// with k' = tap*128 + c, tap = 3p+q (tap-major K permutation).
__global__ void prep_bf16(const float* __restrict__ x, const float* __restrict__ w,
                          unsigned short* __restrict__ xt, unsigned short* __restrict__ wb) {
  int idx = blockIdx.x * blockDim.x + threadIdx.x;
  if (idx < XB_ELEMS) {
    int c    = idx & 127;
    int wcol = (idx >> 7) & 31;
    int h    = (idx >> 12) & 31;
    int n    = idx >> 17;
    xt[idx] = f32_to_bf16_rne(x[((n * C_IN + c) * H_IMG + h) * W_IMG + wcol]);
  } else if (idx < XB_ELEMS + WB_ELEMS) {
    int widx = idx - XB_ELEMS;
    int f    = widx / K_TOT;
    int rem  = widx - f * K_TOT;
    int tap  = rem >> 7;
    int c    = rem & 127;
    wb[widx] = f32_to_bf16_rne(w[(f * C_IN + c) * 9 + tap]);
  }
}

// Implicit-GEMM 3x3 conv, pad=1.  One workgroup = one (n, i) output row:
// 128 filters x 32 columns.  Wave w -> filters [16w, 16w+16).
__global__ __launch_bounds__(256) void conv_wmma(
    const unsigned short* __restrict__ xt,   // bf16 x, NHWC
    const unsigned short* __restrict__ wb,   // bf16 w, [F][tap*128+c]
    const float* __restrict__ bias,
    float* __restrict__ out) {
  // slab[(r*34 + jj)*CP + c] = x[n, c, i-1+r, jj];  jj==32 is the zero pad slot.
  __shared__ __align__(16) unsigned short slab[SLAB_ELEMS];

  const int tid  = threadIdx.x;
  const int wave = tid >> 5;
  const int lane = tid & 31;
  const int l15  = lane & 15;
  const int half = (lane >> 4) & 1;

  const int n = blockIdx.x >> 5;
  const int i = blockIdx.x & 31;

  // Zero the pad column (jj = 32) for each r: 3 * 128 elems.
  if (tid < 96) {
    int r = tid >> 5, l = tid & 31;
    unsigned int* p = (unsigned int*)&slab[(r * 34 + 32) * CP + l * 4];
    p[0] = 0u;
    p[1] = 0u;
  }

  // Async fill of the 3x32xC slab: 48 wave-ops of 2 jj-rows each, 6 per wave.
  // Each lane moves 16 B (8 channels) with global_load_async_to_lds_b128.
  const unsigned long long xbase = (unsigned long long)(size_t)xt;
#pragma unroll
  for (int k = 0; k < 6; ++k) {
    int o  = wave + 8 * k;
    int r  = o >> 4;
    int jj = ((o & 15) << 1) + half;
    int ih = i - 1 + r;
    unsigned lds_off =
        (unsigned)(size_t)&slab[(r * 34 + jj) * CP + l15 * 8];
    if ((unsigned)ih < (unsigned)H_IMG) {
      unsigned long long g =
          xbase + (unsigned long long)((((n * H_IMG + ih) * W_IMG + jj) * C_IN + l15 * 8) * 2);
      asm volatile("global_load_async_to_lds_b128 %0, %1, off"
                   :: "v"(lds_off), "v"(g)
                   : "memory");
    } else {
      u32x4 z = {0u, 0u, 0u, 0u};
      *(u32x4*)&slab[(r * 34 + jj) * CP + l15 * 8] = z;
    }
  }
  asm volatile("s_wait_asynccnt 0" ::: "memory");
  __syncthreads();

  const unsigned short* wrow = wb + (wave * 16 + l15) * K_TOT;  // A row (filter) for this lane

  v8f acc0 = {0.f, 0.f, 0.f, 0.f, 0.f, 0.f, 0.f, 0.f};  // cols 0..15
  v8f acc1 = acc0;                                      // cols 16..31

#pragma unroll
  for (int tap = 0; tap < 9; ++tap) {
    const int r = tap / 3;
    const int s = tap - r * 3;
    // im2col source column for this tap; OOB -> zero slot jj=32
    int jx0 = l15 + s - 1;
    int ji0 = ((unsigned)jx0 > 31u) ? 32 : jx0;
    int jx1 = 16 + l15 + s - 1;
    int ji1 = ((unsigned)jx1 > 31u) ? 32 : jx1;
    const unsigned short* b0 = &slab[(r * 34 + ji0) * CP + half * 8];
    const unsigned short* b1 = &slab[(r * 34 + ji1) * CP + half * 8];
    const unsigned short* wk = wrow + tap * 128;
    __builtin_prefetch(wk + 512, 0, 3);  // global_prefetch weights 4 taps (512 elems) ahead

#pragma unroll
    for (int cb = 0; cb < 4; ++cb) {
      const int co = cb * 32;
      union Frag { u32x4 q[2]; v16bf v; } fa, fb0, fb1;
      // A 16x32 bf16: lanes<16 hold K{0..7,16..23}, lanes>=16 K{8..15,24..31}
      fa.q[0] = *(const u32x4*)(wk + co + half * 8);
      fa.q[1] = *(const u32x4*)(wk + co + 16 + half * 8);
      // B 32x16 bf16: lanes<16 hold K{0..7,16..23}, lanes>=16 K{8..15,24..31}
      fb0.q[0] = *(const u32x4*)(b0 + co);
      fb0.q[1] = *(const u32x4*)(b0 + co + 16);
      fb1.q[0] = *(const u32x4*)(b1 + co);
      fb1.q[1] = *(const u32x4*)(b1 + co + 16);

      acc0 = __builtin_amdgcn_wmma_f32_16x16x32_bf16(false, fa.v, false, fb0.v,
                                                     (short)0, acc0, false, false);
      acc1 = __builtin_amdgcn_wmma_f32_16x16x32_bf16(false, fa.v, false, fb1.v,
                                                     (short)0, acc1, false, false);
    }
  }

  // C/D layout: VGPR v -> M = v (lanes<16) or 8+v (lanes>=16); lane -> N.
#pragma unroll
  for (int v = 0; v < 8; v++) {
    int f = wave * 16 + half * 8 + v;
    float bv = bias[f];
    float* orow = out + (((n * F_OUT + f) * H_IMG + i) * W_IMG);
    orow[l15]      = acc0[v] + bv;
    orow[16 + l15] = acc1[v] + bv;
  }
}

extern "C" void kernel_launch(void* const* d_in, const int* in_sizes, int n_in,
                              void* d_out, int out_size, void* d_ws, size_t ws_size,
                              hipStream_t stream) {
  const float* x    = (const float*)d_in[0];
  const float* w    = (const float*)d_in[1];
  const float* bias = (const float*)d_in[2];
  float* out = (float*)d_out;

  unsigned short* xt = (unsigned short*)d_ws;
  unsigned short* wb = xt + XB_ELEMS;   // 8 MB in, 16B aligned

  int total = XB_ELEMS + WB_ELEMS;
  prep_bf16<<<(total + 255) / 256, 256, 0, stream>>>(x, w, xt, wb);
  conv_wmma<<<N_IMG * H_IMG, 256, 0, stream>>>(xt, wb, bias, out);
}
